// BidirectionalTransformer_48679159333019
// MI455X (gfx1250) — matrix operand order repaired and checked by reading, not compiled, not run on hardware
//
#include <hip/hip_runtime.h>
#include <hip/hip_bf16.h>

// Problem constants (match reference)
#define B_   4
#define L_   1024
#define E_   512
#define H_   8
#define NL_  2
#define M_   2048
#define S_   256
#define R_   2
#define DH_  64
#define LK_  (2 * L_ + 1)
#define LN10000 9.210340371976184f

typedef __attribute__((ext_vector_type(16))) _Float16 v16h;
typedef __attribute__((ext_vector_type(8)))  float    v8f;

union Frag16 {
  v16h v;
  unsigned int u[8];
  _Float16 h[16];
};

__device__ __forceinline__ v8f wmma_f16(v16h a, v16h b, v8f c) {
  // D = A(16x32 f16) x B(32x16 f16) + C(16x16 f32)
  return __builtin_amdgcn_wmma_f32_16x16x32_f16(false, a, false, b, (short)0, c,
                                                false, false);
}

__device__ __forceinline__ unsigned pk2(float x, float y) {
  union { _Float16 h[2]; unsigned u; } t;
  t.h[0] = (_Float16)x;
  t.h[1] = (_Float16)y;
  return t.u;
}

__device__ __forceinline__ float gelu_exact(float x) {
  return 0.5f * x * (1.0f + erff(x * 0.70710678118654752f));
}

// -------------------------------------------------------------------------
// Time embedding
// -------------------------------------------------------------------------
__global__ void temb_kernel(const float* __restrict__ t, float* __restrict__ temb) {
  int i = blockIdx.x * blockDim.x + threadIdx.x;
  if (i >= B_ * E_) return;
  int b = i / E_, c = i % E_;
  const int half = E_ / 2;
  float lg = -LN10000 / (float)(half - 1);
  if (c < half) temb[i] = __sinf(t[b] * __expf((float)c * lg));
  else          temb[i] = __cosf(t[b] * __expf((float)(c - half) * lg));
}

// -------------------------------------------------------------------------
// Assemble h for a direction: shifted token embeds + cond row + posenc
// -------------------------------------------------------------------------
__global__ void build_h_kernel(const int* __restrict__ x, const float* __restrict__ embed,
                               const float* __restrict__ temb, float* __restrict__ h,
                               int dir) {
  int i = blockIdx.x * blockDim.x + threadIdx.x;
  if (i >= B_ * L_ * E_) return;
  int e = i % E_;
  int p = (i / E_) % L_;
  int b = i / (E_ * L_);
  float base;
  if (dir == 0) {
    base = (p == 0) ? temb[b * E_ + e] : embed[(size_t)x[b * L_ + p - 1] * E_ + e];
  } else {
    base = (p == L_ - 1) ? temb[b * E_ + e] : embed[(size_t)x[b * L_ + p + 1] * E_ + e];
  }
  float ang = (float)p * __expf((float)(e & ~1) * (-LN10000 / (float)E_));
  float pe = (e & 1) ? __cosf(ang) : __sinf(ang);
  h[i] = base + pe;
}

__global__ void add_kernel(const float* __restrict__ a, const float* __restrict__ b,
                           float* __restrict__ c, int n) {
  int i = blockIdx.x * blockDim.x + threadIdx.x;
  if (i < n) c[i] = a[i] + b[i];
}

// allv[b, 0] = temb[b]; allv[b, 1..L] = l2rn; allv[b, L+1..2L] = r2ln
__global__ void allv_kernel(const float* __restrict__ temb, const float* __restrict__ l2rn,
                            const float* __restrict__ r2ln, float* __restrict__ allv) {
  int i = blockIdx.x * blockDim.x + threadIdx.x;
  if (i >= B_ * LK_ * E_) return;
  int e = i % E_;
  int p = (i / E_) % LK_;
  int b = i / (E_ * LK_);
  float v;
  if (p == 0)       v = temb[b * E_ + e];
  else if (p <= L_) v = l2rn[((size_t)b * L_ + p - 1) * E_ + e];
  else              v = r2ln[((size_t)b * L_ + p - 1 - L_) * E_ + e];
  allv[i] = v;
}

// -------------------------------------------------------------------------
// LayerNorm: one wave32 per row, optional fused FiLM (fa*y + fb)
// -------------------------------------------------------------------------
__global__ __launch_bounds__(128) void ln_kernel(const float* __restrict__ X,
                                                 const float* __restrict__ s,
                                                 const float* __restrict__ bia,
                                                 const float* __restrict__ film,
                                                 float* __restrict__ Y,
                                                 int rows, int E, int rowsPerBatch) {
  int wave = threadIdx.x >> 5, lane = threadIdx.x & 31;
  int row = blockIdx.x * 4 + wave;
  if (row >= rows) return;
  const float* xr = X + (size_t)row * E;
  float sum = 0.f, sq = 0.f;
  for (int e = lane; e < E; e += 32) {
    float v = xr[e];
    sum += v;
    sq += v * v;
  }
  for (int off = 16; off >= 1; off >>= 1) {
    sum += __shfl_xor(sum, off, 32);
    sq  += __shfl_xor(sq, off, 32);
  }
  float mean = sum / (float)E;
  float var = sq / (float)E - mean * mean;
  float rs = rsqrtf(var + 1e-5f);
  float* yr = Y + (size_t)row * E;
  int fbase = film ? (row / rowsPerBatch) * (2 * E) : 0;
  for (int e = lane; e < E; e += 32) {
    float v = (xr[e] - mean) * rs * s[e] + bia[e];
    if (film) v = film[fbase + e] * v + film[fbase + E + e];
    yr[e] = v;
  }
}

// -------------------------------------------------------------------------
// WMMA GEMM: C[M,N] = epi(alpha*(A[M,K] @ W[K,N] + bias) [+res])
// 128x128 tile per block, 8 waves, BK=32. Requires N % 128 == 0, K % 32 == 0.
// Software-pipelined: double-buffered LDS, next tile prefetched into
// registers before the WMMA block, one barrier per BK step.
// -------------------------------------------------------------------------
__global__ __launch_bounds__(256) void gemm_kernel(const float* __restrict__ A,
                                                   const float* __restrict__ W,
                                                   const float* __restrict__ bias,
                                                   const float* __restrict__ res,
                                                   float* __restrict__ C,
                                                   int Mr, int N, int K,
                                                   float alpha, int act) {
  __shared__ _Float16 sA[2][128][40];   // [buf][row][k]
  __shared__ _Float16 sBt[2][128][40];  // [buf][n][k]  (W tile transposed)
  int tid = threadIdx.x;
  int wave = tid >> 5, lane = tid & 31;
  int hi = lane >> 4, ln = lane & 15;
  int am = wave & 3;   // M sub-block (32 rows)
  int an = wave >> 2;  // N sub-block (64 cols)
  int bm = blockIdx.y * 128, bn = blockIdx.x * 128;

  // Staging assignments (uniform per thread)
  int sr = tid >> 1;                 // A row 0..127
  int sk0 = (tid & 1) << 4;          // A k-offset 0/16
  int wn = tid & 127;                // W col 0..127
  int wk0 = (tid >> 7) << 4;         // W k-offset 0/16
  int grow = bm + sr;

  auto loadA = [&](int k0, uint4& u0, uint4& u1) {
    if (grow < Mr) {
      const float4* src = (const float4*)(A + (size_t)grow * K + k0 + sk0);
      float4 f0 = src[0], f1 = src[1], f2 = src[2], f3 = src[3];
      u0 = {pk2(f0.x, f0.y), pk2(f0.z, f0.w), pk2(f1.x, f1.y), pk2(f1.z, f1.w)};
      u1 = {pk2(f2.x, f2.y), pk2(f2.z, f2.w), pk2(f3.x, f3.y), pk2(f3.z, f3.w)};
    } else {
      u0 = {0u, 0u, 0u, 0u};
      u1 = {0u, 0u, 0u, 0u};
    }
  };
  auto loadW = [&](int k0, uint4& u0, uint4& u1) {
    const float* wp = W + (size_t)(k0 + wk0) * N + bn + wn;
    float w[16];
#pragma unroll
    for (int j = 0; j < 16; ++j) w[j] = wp[(size_t)j * N];
    u0 = {pk2(w[0], w[1]), pk2(w[2], w[3]), pk2(w[4], w[5]), pk2(w[6], w[7])};
    u1 = {pk2(w[8], w[9]), pk2(w[10], w[11]), pk2(w[12], w[13]), pk2(w[14], w[15])};
  };

  v8f zero = {0.f, 0.f, 0.f, 0.f, 0.f, 0.f, 0.f, 0.f};
  v8f acc[2][4];
#pragma unroll
  for (int s = 0; s < 2; ++s)
#pragma unroll
    for (int t = 0; t < 4; ++t) acc[s][t] = zero;

  uint4 ra0, ra1, rb0, rb1;
  loadA(0, ra0, ra1);
  loadW(0, rb0, rb1);

  int nsteps = K >> 5;
  int cur = 0;
  for (int i = 0; i < nsteps; ++i) {
    *(uint4*)&sA[cur][sr][sk0]       = ra0;
    *(uint4*)&sA[cur][sr][sk0 + 8]   = ra1;
    *(uint4*)&sBt[cur][wn][wk0]      = rb0;
    *(uint4*)&sBt[cur][wn][wk0 + 8]  = rb1;
    __syncthreads();
    if (i + 1 < nsteps) {   // prefetch next tile; overlaps the WMMAs below
      loadA((i + 1) << 5, ra0, ra1);
      loadW((i + 1) << 5, rb0, rb1);
    }
    Frag16 a[2];
#pragma unroll
    for (int s = 0; s < 2; ++s) {
#pragma unroll
      for (int r = 0; r < 8; ++r) {
        int kb = ((r & 4) ? 16 : 0) + (hi << 3) + ((r & 3) << 1);
        a[s].u[r] = *(const unsigned int*)&sA[cur][am * 32 + s * 16 + ln][kb];
      }
    }
#pragma unroll
    for (int t = 0; t < 4; ++t) {
      Frag16 bf;
#pragma unroll
      for (int v = 0; v < 8; ++v)
        bf.u[v] = *(const unsigned int*)&sBt[cur][an * 64 + t * 16 + ln][((v & 4) ? 16 : 0) + (hi << 3) + ((v & 3) << 1)];
#pragma unroll
      for (int s = 0; s < 2; ++s) acc[s][t] = wmma_f16(a[s].v, bf.v, acc[s][t]);
    }
    cur ^= 1;
  }

#pragma unroll
  for (int t = 0; t < 4; ++t) {
    int col = bn + an * 64 + t * 16 + ln;
    float bv = bias ? bias[col] : 0.f;   // hoisted: independent of s, r
#pragma unroll
    for (int s = 0; s < 2; ++s) {
#pragma unroll
      for (int r = 0; r < 8; ++r) {
        int row = bm + am * 32 + s * 16 + (hi << 3) + r;
        if (row < Mr) {
          float v = (acc[s][t][r] + bv) * alpha;
          if (act) v = gelu_exact(v);
          if (res) v += res[(size_t)row * N + col];
          C[(size_t)row * N + col] = v;
        }
      }
    }
  }
}

// -------------------------------------------------------------------------
// Flash attention: one wave per 16 query rows, 32-key blocks, online softmax.
// Q pre-scaled by 1/sqrt(Dh). mode: 0 = k<=q, 1 = k>=q, 2 = joint readout.
// -------------------------------------------------------------------------
__device__ __forceinline__ bool mask_ok(int mode, int q, int k, int Lk) {
  if (k >= Lk) return false;
  if (mode == 0) return k <= q;
  if (mode == 1) return k >= q;
  if (k == 0) return true;                 // temb key
  if (k <= L_) return (k - 1) > q;         // l2r segment
  return (k - 1 - L_) < q;                 // r2l segment
}

__global__ __launch_bounds__(128) void attn_kernel(const float* __restrict__ Q,
                                                   const float* __restrict__ Km,
                                                   const float* __restrict__ Vm,
                                                   float* __restrict__ O,
                                                   int Lk, int kvStride, int mode) {
  __shared__ _Float16 sK[32][72];      // [key][feat]
  __shared__ _Float16 sVt[64][40];     // [feat][key]
  __shared__ _Float16 sP[4][16][40];   // per-wave probability tile

  int tid = threadIdx.x;
  int wave = tid >> 5, lane = tid & 31;
  int hi = lane >> 4, ln = lane & 15;
  int b = blockIdx.z, h = blockIdx.y;
  int q0 = blockIdx.x * 64 + wave * 16;

  // staging assignment: one key per thread, 16 contiguous feats
  int skey = tid >> 2;
  int sf0 = (tid & 3) << 4;

  // Q fragments: two 16x32 chunks covering Dh=64
  Frag16 qf[2];
  const float* qp = Q + (size_t)b * (L_ * E_) + (size_t)(q0 + ln) * E_ + h * DH_;
#pragma unroll
  for (int c = 0; c < 2; ++c) {
#pragma unroll
    for (int r = 0; r < 8; ++r) {
      int kb = c * 32 + ((r & 4) ? 16 : 0) + (hi << 3) + ((r & 3) << 1);
      float2 two = *(const float2*)(qp + kb);
      qf[c].h[2 * r]     = (_Float16)two.x;
      qf[c].h[2 * r + 1] = (_Float16)two.y;
    }
  }

  float m_[8], l_[8];
  v8f zero = {0.f, 0.f, 0.f, 0.f, 0.f, 0.f, 0.f, 0.f};
  v8f o_[4];
#pragma unroll
  for (int r = 0; r < 8; ++r) { m_[r] = -1e30f; l_[r] = 0.f; }
#pragma unroll
  for (int t = 0; t < 4; ++t) o_[t] = zero;

  for (int kb0 = 0; kb0 < Lk; kb0 += 32) {
    // ---- Stage K (natural) / V (transposed); guard uniform per thread ----
    {
      float kf[16], vf[16];
      if (kb0 + skey < Lk) {
        size_t base = (size_t)b * kvStride + (size_t)(kb0 + skey) * E_ + h * DH_ + sf0;
        const float4* kp = (const float4*)(Km + base);
        const float4* vp = (const float4*)(Vm + base);
#pragma unroll
        for (int j = 0; j < 4; ++j) {
          float4 a = kp[j], v = vp[j];
          kf[4 * j + 0] = a.x; kf[4 * j + 1] = a.y; kf[4 * j + 2] = a.z; kf[4 * j + 3] = a.w;
          vf[4 * j + 0] = v.x; vf[4 * j + 1] = v.y; vf[4 * j + 2] = v.z; vf[4 * j + 3] = v.w;
        }
      } else {
#pragma unroll
        for (int j = 0; j < 16; ++j) { kf[j] = 0.f; vf[j] = 0.f; }
      }
      uint4 u0 = {pk2(kf[0], kf[1]), pk2(kf[2], kf[3]), pk2(kf[4], kf[5]), pk2(kf[6], kf[7])};
      uint4 u1 = {pk2(kf[8], kf[9]), pk2(kf[10], kf[11]), pk2(kf[12], kf[13]), pk2(kf[14], kf[15])};
      *(uint4*)&sK[skey][sf0] = u0;
      *(uint4*)&sK[skey][sf0 + 8] = u1;
#pragma unroll
      for (int j = 0; j < 16; ++j) sVt[sf0 + j][skey] = (_Float16)vf[j];
    }
    // Prefetch next key block into L2/L0 (global_prefetch_b8); overlaps
    // the softmax + WMMA work below with the next block's HBM fetch.
    if (kb0 + 32 + skey < Lk) {
      size_t pbase = (size_t)b * kvStride + (size_t)(kb0 + 32 + skey) * E_ + h * DH_ + sf0;
      __builtin_prefetch(Km + pbase, 0, 3);
      __builtin_prefetch(Vm + pbase, 0, 3);
    }
    __syncthreads();

    // S = Q @ K^T (4 WMMAs)
    v8f s0 = zero, s1 = zero;
#pragma unroll
    for (int c = 0; c < 2; ++c) {
      Frag16 b0, b1;
#pragma unroll
      for (int v = 0; v < 8; ++v) {
        int kf2 = c * 32 + ((v & 4) ? 16 : 0) + (hi << 3) + ((v & 3) << 1);
        b0.u[v] = *(const unsigned int*)&sK[ln][kf2];
        b1.u[v] = *(const unsigned int*)&sK[16 + ln][kf2];
      }
      s0 = wmma_f16(qf[c].v, b0.v, s0);
      s1 = wmma_f16(qf[c].v, b1.v, s1);
    }

    // Mask + online softmax
#pragma unroll
    for (int r = 0; r < 8; ++r) {
      int qi = q0 + (hi << 3) + r;
      bool a0 = mask_ok(mode, qi, kb0 + ln, Lk);
      bool a1 = mask_ok(mode, qi, kb0 + 16 + ln, Lk);
      float e0 = a0 ? s0[r] : -1e30f;
      float e1 = a1 ? s1[r] : -1e30f;
      float rmx = fmaxf(e0, e1);
#pragma unroll
      for (int off = 8; off >= 1; off >>= 1) rmx = fmaxf(rmx, __shfl_xor(rmx, off, 32));
      float mnew = fmaxf(m_[r], rmx);
      float scl = __expf(m_[r] - mnew);
      float p0 = a0 ? __expf(e0 - mnew) : 0.f;
      float p1 = a1 ? __expf(e1 - mnew) : 0.f;
      float rsum = p0 + p1;
#pragma unroll
      for (int off = 8; off >= 1; off >>= 1) rsum += __shfl_xor(rsum, off, 32);
      l_[r] = l_[r] * scl + rsum;
      m_[r] = mnew;
#pragma unroll
      for (int t = 0; t < 4; ++t) o_[t][r] *= scl;
      sP[wave][(hi << 3) + r][ln]      = (_Float16)p0;
      sP[wave][(hi << 3) + r][16 + ln] = (_Float16)p1;
    }
    __syncthreads();

    // O += P @ V (4 WMMAs)
    Frag16 ap;
#pragma unroll
    for (int r = 0; r < 8; ++r) {
      int kb = ((r & 4) ? 16 : 0) + (hi << 3) + ((r & 3) << 1);
      ap.u[r] = *(const unsigned int*)&sP[wave][ln][kb];
    }
#pragma unroll
    for (int t = 0; t < 4; ++t) {
      Frag16 bv;
#pragma unroll
      for (int v = 0; v < 8; ++v)
        bv.u[v] = *(const unsigned int*)&sVt[t * 16 + ln][((v & 4) ? 16 : 0) + (hi << 3) + ((v & 3) << 1)];
      o_[t] = wmma_f16(ap.v, bv.v, o_[t]);
    }
    __syncthreads();
  }

#pragma unroll
  for (int t = 0; t < 4; ++t) {
#pragma unroll
    for (int r = 0; r < 8; ++r) {
      int row = q0 + (hi << 3) + r;
      int col = h * DH_ + t * 16 + ln;
      float denom = (l_[r] > 0.f) ? l_[r] : 1.f;
      O[(size_t)b * (L_ * E_) + (size_t)row * E_ + col] = o_[t][r] / denom;
    }
  }
}

// -------------------------------------------------------------------------
// Host orchestration
// -------------------------------------------------------------------------
extern "C" void kernel_launch(void* const* d_in, const int* in_sizes, int n_in,
                              void* d_out, int out_size, void* d_ws, size_t ws_size,
                              hipStream_t stream) {
  (void)in_sizes; (void)n_in; (void)out_size; (void)ws_size;
  const int*   x        = (const int*)d_in[0];
  const float* t        = (const float*)d_in[1];
  const float* embed    = (const float*)d_in[2];
  const float* attn_W   = (const float*)d_in[3];
  const float* attn_b   = (const float*)d_in[4];
  const float* ln_s     = (const float*)d_in[5];
  const float* ln_b     = (const float*)d_in[6];
  const float* mlp_W1   = (const float*)d_in[7];
  const float* mlp_b1   = (const float*)d_in[8];
  const float* mlp_W2   = (const float*)d_in[9];
  const float* mlp_b2   = (const float*)d_in[10];
  const float* ca_Wq    = (const float*)d_in[11];
  const float* ca_Wk    = (const float*)d_in[12];
  const float* ca_bk    = (const float*)d_in[13];
  const float* ca_Wv    = (const float*)d_in[14];
  const float* ca_bv    = (const float*)d_in[15];
  const float* ca_Wo    = (const float*)d_in[16];
  const float* ca_bo    = (const float*)d_in[17];
  const float* ro_ln1_s = (const float*)d_in[18];
  const float* ro_ln1_b = (const float*)d_in[19];
  const float* ro_ln2_s = (const float*)d_in[20];
  const float* ro_ln2_b = (const float*)d_in[21];
  const float* in_W     = (const float*)d_in[22];
  const float* in_b     = (const float*)d_in[23];
  const float* tmlp_W1  = (const float*)d_in[24];
  const float* tmlp_b1  = (const float*)d_in[25];
  const float* tmlp_W2  = (const float*)d_in[26];
  const float* tmlp_b2  = (const float*)d_in[27];
  const float* res_W1   = (const float*)d_in[28];
  const float* res_b1   = (const float*)d_in[29];
  const float* res_W2   = (const float*)d_in[30];
  const float* res_b2   = (const float*)d_in[31];
  const float* res_ln_s = (const float*)d_in[32];
  const float* res_ln_b = (const float*)d_in[33];
  const float* film_W   = (const float*)d_in[34];
  const float* film_b   = (const float*)d_in[35];
  const float* out_W    = (const float*)d_in[36];
  const float* out_b    = (const float*)d_in[37];

  float* ws = (float*)d_ws;
  size_t off = 0;
  auto alloc = [&](size_t n) { float* p = ws + off; off += n; return p; };
  const size_t BLE = (size_t)B_ * L_ * E_;

  float* tembBuf = alloc((size_t)B_ * E_);
  float* h0      = alloc(BLE);
  float* h1      = alloc(BLE);
  float* zbuf    = alloc(BLE);
  float* qbuf    = alloc(BLE);
  float* kbuf    = alloc(BLE);
  float* vbuf    = alloc(BLE);
  float* obuf    = alloc(BLE);
  float* gbuf    = alloc((size_t)B_ * L_ * M_);
  float* inputs  = alloc(BLE);
  float* l2rn    = alloc(BLE);
  float* r2ln    = alloc(BLE);
  float* qsum    = alloc(BLE);
  float* allv    = alloc((size_t)B_ * LK_ * E_);
  float* kall    = alloc((size_t)B_ * LK_ * E_);
  float* vall    = alloc((size_t)B_ * LK_ * E_);
  float* xat     = alloc(BLE);
  float* xat2    = alloc(BLE);
  float* tg      = alloc((size_t)B_ * M_);
  float* t4      = alloc((size_t)B_ * 4 * E_);
  float* filmBuf = alloc((size_t)B_ * 4 * E_);
  float* hbuf    = alloc((size_t)B_ * L_ * 2 * E_);
  float* tmpBuf  = alloc((size_t)B_ * L_ * 2 * E_);

  const int nBL = B_ * L_;
  const int nKV = B_ * LK_;

  auto gemm = [&](const float* A, const float* W, const float* bias, const float* res,
                  float* C, int M, int N, int K, float alpha, int act) {
    dim3 grid(N / 128, (M + 127) / 128);
    gemm_kernel<<<grid, dim3(256), 0, stream>>>(A, W, bias, res, C, M, N, K, alpha, act);
  };
  auto lnorm = [&](const float* X, const float* s, const float* bb, const float* film,
                   float* Y, int rows, int E) {
    ln_kernel<<<dim3((rows + 3) / 4), dim3(128), 0, stream>>>(X, s, bb, film, Y, rows, E, L_);
  };

  temb_kernel<<<dim3((B_ * E_ + 255) / 256), dim3(256), 0, stream>>>(t, tembBuf);
  build_h_kernel<<<dim3((int)((BLE + 255) / 256)), dim3(256), 0, stream>>>(x, embed, tembBuf, h0, 0);
  build_h_kernel<<<dim3((int)((BLE + 255) / 256)), dim3(256), 0, stream>>>(x, embed, tembBuf, h1, 1);

  float* hs[2] = {h0, h1};
  for (int d = 0; d < 2; ++d) {
    float* h = hs[d];
    for (int l = 0; l < NL_; ++l) {
      size_t wOff  = (size_t)(d * NL_ + l) * 4 * E_ * E_;
      size_t bOff  = (size_t)(d * NL_ + l) * 4 * E_;
      size_t lnOff = (size_t)(d * NL_ + l) * 2 * E_;
      lnorm(h, ln_s + lnOff, ln_b + lnOff, nullptr, zbuf, nBL, E_);
      gemm(zbuf, attn_W + wOff + 0 * (size_t)E_ * E_, attn_b + bOff + 0 * E_, nullptr,
           qbuf, nBL, E_, E_, 0.125f, 0);                       // q / sqrt(Dh)
      gemm(zbuf, attn_W + wOff + 1 * (size_t)E_ * E_, attn_b + bOff + 1 * E_, nullptr,
           kbuf, nBL, E_, E_, 1.f, 0);
      gemm(zbuf, attn_W + wOff + 2 * (size_t)E_ * E_, attn_b + bOff + 2 * E_, nullptr,
           vbuf, nBL, E_, E_, 1.f, 0);
      attn_kernel<<<dim3(L_ / 64, H_, B_), dim3(128), 0, stream>>>(
          qbuf, kbuf, vbuf, obuf, L_, L_ * E_, d);              // d==0 causal, d==1 anti
      gemm(obuf, attn_W + wOff + 3 * (size_t)E_ * E_, attn_b + bOff + 3 * E_, h,
           h, nBL, E_, E_, 1.f, 0);                             // h += o@W3+b3
      lnorm(h, ln_s + lnOff + E_, ln_b + lnOff + E_, nullptr, zbuf, nBL, E_);
      size_t m1Off = (size_t)(d * NL_ + l) * E_ * M_;
      size_t m2Off = (size_t)(d * NL_ + l) * M_ * E_;
      gemm(zbuf, mlp_W1 + m1Off, mlp_b1 + (size_t)(d * NL_ + l) * M_, nullptr,
           gbuf, nBL, M_, E_, 1.f, 1);                          // gelu
      gemm(gbuf, mlp_W2 + m2Off, mlp_b2 + (size_t)(d * NL_ + l) * E_, h,
           h, nBL, E_, M_, 1.f, 0);                             // h += g@W2+b2
    }
  }

  add_kernel<<<dim3((int)((BLE + 255) / 256)), dim3(256), 0, stream>>>(h0, h1, inputs, (int)BLE);
  lnorm(h0, ro_ln1_s, ro_ln1_b, nullptr, l2rn, nBL, E_);
  lnorm(h1, ro_ln2_s, ro_ln2_b, nullptr, r2ln, nBL, E_);
  add_kernel<<<dim3((int)((BLE + 255) / 256)), dim3(256), 0, stream>>>(l2rn, r2ln, qsum, (int)BLE);
  gemm(qsum, ca_Wq, nullptr, nullptr, qbuf, nBL, E_, E_, 0.125f, 0);
  allv_kernel<<<dim3((B_ * LK_ * E_ + 255) / 256), dim3(256), 0, stream>>>(tembBuf, l2rn, r2ln, allv);
  gemm(allv, ca_Wk, ca_bk, nullptr, kall, nKV, E_, E_, 1.f, 0);
  gemm(allv, ca_Wv, ca_bv, nullptr, vall, nKV, E_, E_, 1.f, 0);
  attn_kernel<<<dim3(L_ / 64, H_, B_), dim3(128), 0, stream>>>(
      qbuf, kall, vall, xat, LK_, LK_ * E_, 2);                 // joint mask
  gemm(xat, ca_Wo, ca_bo, inputs, xat2, nBL, E_, E_, 1.f, 0);

  gemm(tembBuf, tmlp_W1, tmlp_b1, nullptr, tg, B_, M_, E_, 1.f, 1);
  gemm(tg, tmlp_W2, tmlp_b2, nullptr, t4, B_, 4 * E_, M_, 1.f, 0);
  gemm(xat2, in_W, in_b, nullptr, hbuf, nBL, 2 * E_, E_, 1.f, 0);

  for (int i = 0; i < R_; ++i) {
    gemm(t4, film_W + (size_t)i * 4 * E_ * 4 * E_, film_b + (size_t)i * 4 * E_, nullptr,
         filmBuf, B_, 4 * E_, 4 * E_, 1.f, 0);
    gemm(hbuf, res_W1 + (size_t)i * 2 * E_ * M_, res_b1 + (size_t)i * M_, nullptr,
         gbuf, nBL, M_, 2 * E_, 1.f, 1);                        // gelu
    gemm(gbuf, res_W2 + (size_t)i * M_ * 2 * E_, res_b2 + (size_t)i * 2 * E_, hbuf,
         tmpBuf, nBL, 2 * E_, M_, 1.f, 0);                      // h + z
    lnorm(tmpBuf, res_ln_s + (size_t)i * 2 * E_, res_ln_b + (size_t)i * 2 * E_,
          filmBuf, hbuf, nBL, 2 * E_);                          // LN + FiLM
  }
  gemm(hbuf, out_W, out_b, nullptr, (float*)d_out, nBL, S_, 2 * E_, 1.f, 0);
}
